// FCPlanenet_63788854280446
// MI455X (gfx1250) — matrix operand order repaired and checked by reading, not compiled, not run on hardware
//
#include <hip/hip_runtime.h>
#include <hip/hip_bf16.h>

typedef __attribute__((ext_vector_type(16))) _Float16 v16h;
typedef __attribute__((ext_vector_type(8)))  _Float16 v8h;
typedef __attribute__((ext_vector_type(8)))  float    v8f;

#define BATCH 32
#define TPTS  8192
#define HID   128
#define POSD  256

// ---------------------------------------------------------------------------
// CDNA5 v_wmma_f32_16x16x32_f16 fragment layouts (ISA 7.12.2):
//   A (16x32): lane<16 -> M=lane,   K in {0..7, 16..23};
//              lane>=16 -> M=lane-16, K in {8..15, 24..31}
//     => per lane, two contiguous 8-half runs: [half*8 .. +7] and [16+half*8 .. +7]
//   B (32x16): lane = half*16 + n holds K = half*8 + i (i<8), 16+half*8 + i-8 (i>=8)
//   C/D: VGPR j -> row M = j + 8*half, col N = lane&15
// ---------------------------------------------------------------------------

__device__ __forceinline__ v16h load_a_frag(const _Float16* __restrict__ rowbase,
                                            int half, int r) {
  const _Float16* row = rowbase + (size_t)r * HID;
  v8h lo = *(const v8h*)(row + half * 8);        // K = half*8 .. half*8+7
  v8h hi = *(const v8h*)(row + 16 + half * 8);   // K = 16+half*8 .. +7
  v16h a;
#pragma unroll
  for (int i = 0; i < 8; ++i) { a[i] = lo[i]; a[i + 8] = hi[i]; }
  return a;
}

// B fragments pre-packed so each lane's 16 halves are contiguous (32 B):
// packed[ ((kc*8 + nt)*32 + lane)*16 + i ]
__device__ __forceinline__ v16h load_b_frag_packed(const _Float16* __restrict__ hW,
                                                   int kc, int nt, int lane) {
  return *(const v16h*)(hW + (((size_t)(kc * 8 + nt)) * 32 + lane) * 16);
}

// ---- weight convert + fragment-order repack (f32 [K][128] -> packed f16) ----
__global__ void cvt_pack(const float* __restrict__ W, _Float16* __restrict__ out, int K) {
  int idx = blockIdx.x * blockDim.x + threadIdx.x;
  if (idx >= K * HID) return;
  int k = idx / HID, ng = idx % HID;
  int kc = k >> 5, kk = k & 31;
  int nt = ng >> 4, n = ng & 15;
  int lane = ((kk >> 3) & 1) * 16 + n;  // half*16 + n
  int i = (kk & 7) + (kk >> 4) * 8;
  out[(((size_t)(kc * 8 + nt)) * 32 + lane) * 16 + i] = (_Float16)W[idx];
}

// ---- Kernel A: z = relu( relu(p@Wpos+bpos) @ W0 + b0 ), pool max(z) ----
// 128 threads = 4 waves x 16 points; grid (T/64, B)
__global__ void kernelA(const float* __restrict__ p, const float* __restrict__ Wpos,
                        const float* __restrict__ bpos, const _Float16* __restrict__ hW0,
                        const float* __restrict__ b0, _Float16* __restrict__ z,
                        unsigned* __restrict__ pooled) {
  __shared__ float    sWpos[3 * POSD];
  __shared__ float    sbpos[POSD];
  __shared__ unsigned smax[HID];
  __shared__ _Float16 stg[4][16][HID];   // per-wave private D staging

  int b = blockIdx.y;
  int wave = threadIdx.x >> 5, lane = threadIdx.x & 31;
  int half = lane >> 4, r = lane & 15;
  int t0 = blockIdx.x * 64 + wave * 16;

  for (int t = threadIdx.x; t < 3 * POSD; t += blockDim.x) sWpos[t] = Wpos[t];
  for (int t = threadIdx.x; t < POSD; t += blockDim.x) sbpos[t] = bpos[t];
  if (threadIdx.x < HID) smax[threadIdx.x] = 0u;
  __syncthreads();

  // pos-encoding computed straight into A-fragment layout (K = 256 -> 8 chunks)
  const float* pr = p + ((size_t)b * TPTS + t0 + r) * 3;
  float p0 = pr[0], p1 = pr[1], p2 = pr[2];
  v16h afr[8];
#pragma unroll
  for (int kc = 0; kc < 8; ++kc) {
    int kbase = kc * 32 + half * 8;
#pragma unroll
    for (int i = 0; i < 8; ++i) {
      int k0 = kbase + i;        // elements 0..7
      int k1 = kbase + 16 + i;   // elements 8..15
      float v0 = p0 * sWpos[k0] + p1 * sWpos[POSD + k0] + p2 * sWpos[2 * POSD + k0] + sbpos[k0];
      float v1 = p0 * sWpos[k1] + p1 * sWpos[POSD + k1] + p2 * sWpos[2 * POSD + k1] + sbpos[k1];
      afr[kc][i]     = (_Float16)(v0 > 0.f ? v0 : 0.f);
      afr[kc][i + 8] = (_Float16)(v1 > 0.f ? v1 : 0.f);
    }
  }

  _Float16* zb = z + ((size_t)b * TPTS + t0) * HID;
#pragma unroll 2
  for (int nt = 0; nt < 8; ++nt) {
    v8f acc = {};
#pragma unroll
    for (int kc = 0; kc < 8; ++kc) {
      v16h bf = load_b_frag_packed(hW0, kc, nt, lane);
      acc = __builtin_amdgcn_wmma_f32_16x16x32_f16(false, afr[kc], false, bf,
                                                   (short)0, acc, false, false);
    }
    int n = nt * 16 + r;
    float bias = b0[n];
    float vmax = 0.f;
#pragma unroll
    for (int j = 0; j < 8; ++j) {
      float v = acc[j] + bias;
      v = v > 0.f ? v : 0.f;
      vmax = v > vmax ? v : vmax;
      stg[wave][j + half * 8][n] = (_Float16)v;
    }
    atomicMax(&smax[n], __float_as_uint(vmax));  // values >= 0: uint order == float order
  }
  // coalesced copy-out: 16 rows x 128 cols are contiguous in z
  {
    const v8h* src = (const v8h*)&stg[wave][0][0];
    v8h* dst = (v8h*)zb;
#pragma unroll
    for (int c = lane; c < 16 * HID / 8; c += 32) dst[c] = src[c];
  }
  __syncthreads();
  if (threadIdx.x < HID) atomicMax(&pooled[(size_t)b * HID + threadIdx.x], smax[threadIdx.x]);
}

// ---- per-batch bias: c = pooled_relu @ W_bot + b (f32, exact weights) ----
__global__ void cvec_kernel(const float* __restrict__ pooled, const float* __restrict__ Wfull,
                            const float* __restrict__ bias, float* __restrict__ c) {
  int b = blockIdx.x, n = threadIdx.x;
  float s = bias[n];
  const float* pb = pooled + b * HID;
  for (int k = 0; k < HID; ++k) s += pb[k] * Wfull[(size_t)(HID + k) * HID + n];
  c[b * HID + n] = s;
}

// ---- Kernel B: z = relu( z @ W_top + c ), in place; pool max(z) ----
// 256 threads = 8 waves x 16 points; grid (T/128, B)
__global__ void kernelB(const _Float16* __restrict__ hW, const float* __restrict__ cvec,
                        _Float16* __restrict__ z, unsigned* __restrict__ pooled) {
  __shared__ unsigned smax[HID];
  __shared__ _Float16 stg[8][16][HID];   // per-wave private D staging (32 KB)

  int b = blockIdx.y;
  int wave = threadIdx.x >> 5, lane = threadIdx.x & 31;
  int half = lane >> 4, r = lane & 15;
  int t0 = blockIdx.x * 128 + wave * 16;
  if (threadIdx.x < HID) smax[threadIdx.x] = 0u;
  __syncthreads();

  _Float16* zb = z + ((size_t)b * TPTS + t0) * HID;
  v16h afr[4];
#pragma unroll
  for (int kc = 0; kc < 4; ++kc)
    afr[kc] = load_a_frag(zb + kc * 32, half, r);   // rows owned exclusively by this wave

  const float* cb = cvec + b * HID;
#pragma unroll
  for (int nt = 0; nt < 8; ++nt) {
    v8f acc = {};
#pragma unroll
    for (int kc = 0; kc < 4; ++kc) {
      v16h bf = load_b_frag_packed(hW, kc, nt, lane);
      acc = __builtin_amdgcn_wmma_f32_16x16x32_f16(false, afr[kc], false, bf,
                                                   (short)0, acc, false, false);
    }
    int n = nt * 16 + r;
    float bias = cb[n];
    float vmax = 0.f;
#pragma unroll
    for (int j = 0; j < 8; ++j) {
      float v = acc[j] + bias;
      v = v > 0.f ? v : 0.f;
      vmax = v > vmax ? v : vmax;
      stg[wave][j + half * 8][n] = (_Float16)v;
    }
    atomicMax(&smax[n], __float_as_uint(vmax));
  }
  {
    const v8h* src = (const v8h*)&stg[wave][0][0];
    v8h* dst = (v8h*)zb;   // in-place: this wave's reads completed (WMMA deps)
#pragma unroll
    for (int c = lane; c < 16 * HID / 8; c += 32) dst[c] = src[c];
  }
  __syncthreads();
  if (threadIdx.x < HID) atomicMax(&pooled[(size_t)b * HID + threadIdx.x], smax[threadIdx.x]);
}

// ---- head MLP on [B,128]: relu at consumption each layer, no relu on output ----
__global__ void head_kernel(const float* __restrict__ pooled3,
                            const float* __restrict__ Wc, const float* __restrict__ bc,
                            const float* __restrict__ Wm0, const float* __restrict__ bm0,
                            const float* __restrict__ Wm1, const float* __restrict__ bm1,
                            const float* __restrict__ Wm2, const float* __restrict__ bm2,
                            const float* __restrict__ Wp, const float* __restrict__ bp,
                            float* __restrict__ out) {
  __shared__ float x[HID];
  int b = blockIdx.x, n = threadIdx.x;
  x[n] = pooled3[b * HID + n];
  __syncthreads();
  const float* Ws[4] = {Wc, Wm0, Wm1, Wm2};
  const float* bs[4] = {bc, bm0, bm1, bm2};
  for (int L = 0; L < 4; ++L) {
    float s = bs[L][n];
    for (int k = 0; k < HID; ++k) {
      float xv = x[k]; xv = xv > 0.f ? xv : 0.f;
      s += xv * Ws[L][(size_t)k * HID + n];
    }
    __syncthreads();
    x[n] = s;
    __syncthreads();
  }
  if (n < 9) {
    float s = bp[n];
    for (int k = 0; k < HID; ++k) {
      float xv = x[k]; xv = xv > 0.f ? xv : 0.f;
      s += xv * Wp[k * 9 + n];
    }
    out[b * 9 + n] = s;
  }
}

extern "C" void kernel_launch(void* const* d_in, const int* in_sizes, int n_in,
                              void* d_out, int out_size, void* d_ws, size_t ws_size,
                              hipStream_t stream) {
  (void)in_sizes; (void)n_in; (void)out_size; (void)ws_size;
  const float* p    = (const float*)d_in[0];
  const float* Wpos = (const float*)d_in[1];
  const float* bpos = (const float*)d_in[2];
  const float* W0   = (const float*)d_in[3];
  const float* b0   = (const float*)d_in[4];
  const float* W1   = (const float*)d_in[5];
  const float* b1   = (const float*)d_in[6];
  const float* W2   = (const float*)d_in[7];
  const float* b2   = (const float*)d_in[8];
  const float* W3   = (const float*)d_in[9];
  const float* b3   = (const float*)d_in[10];
  const float* Wc   = (const float*)d_in[11];
  const float* bc   = (const float*)d_in[12];
  const float* Wm0  = (const float*)d_in[13];
  const float* bm0  = (const float*)d_in[14];
  const float* Wm1  = (const float*)d_in[15];
  const float* bm1  = (const float*)d_in[16];
  const float* Wm2  = (const float*)d_in[17];
  const float* bm2  = (const float*)d_in[18];
  const float* Wp   = (const float*)d_in[19];
  const float* bp   = (const float*)d_in[20];

  char* ws = (char*)d_ws;
  size_t off = 0;
  auto carve = [&](size_t bytes) -> void* {
    void* r = ws + off;
    off += bytes;
    off = (off + 255) & ~(size_t)255;
    return r;
  };
  _Float16* z   = (_Float16*)carve((size_t)BATCH * TPTS * HID * sizeof(_Float16)); // 64 MB, L2-resident
  _Float16* hW0 = (_Float16*)carve((size_t)POSD * HID * sizeof(_Float16));
  _Float16* hW1 = (_Float16*)carve((size_t)HID * HID * sizeof(_Float16));
  _Float16* hW2 = (_Float16*)carve((size_t)HID * HID * sizeof(_Float16));
  _Float16* hW3 = (_Float16*)carve((size_t)HID * HID * sizeof(_Float16));
  unsigned* pooled = (unsigned*)carve((size_t)4 * BATCH * HID * sizeof(unsigned));
  float* c1 = (float*)carve((size_t)BATCH * HID * sizeof(float));
  float* c2 = (float*)carve((size_t)BATCH * HID * sizeof(float));
  float* c3 = (float*)carve((size_t)BATCH * HID * sizeof(float));

  hipMemsetAsync(pooled, 0, (size_t)4 * BATCH * HID * sizeof(unsigned), stream);
  // pack weights into WMMA B-fragment order (f16)
  cvt_pack<<<(POSD * HID + 255) / 256, 256, 0, stream>>>(W0, hW0, POSD);
  cvt_pack<<<(HID * HID + 255) / 256, 256, 0, stream>>>(W1, hW1, HID);  // top half of W1
  cvt_pack<<<(HID * HID + 255) / 256, 256, 0, stream>>>(W2, hW2, HID);
  cvt_pack<<<(HID * HID + 255) / 256, 256, 0, stream>>>(W3, hW3, HID);

  dim3 gA(TPTS / 64, BATCH);
  kernelA<<<gA, 128, 0, stream>>>(p, Wpos, bpos, hW0, b0, z, pooled + 0 * BATCH * HID);

  dim3 gB(TPTS / 128, BATCH);
  cvec_kernel<<<BATCH, HID, 0, stream>>>((const float*)(pooled + 0 * BATCH * HID), W1, b1, c1);
  kernelB<<<gB, 256, 0, stream>>>(hW1, c1, z, pooled + 1 * BATCH * HID);

  cvec_kernel<<<BATCH, HID, 0, stream>>>((const float*)(pooled + 1 * BATCH * HID), W2, b2, c2);
  kernelB<<<gB, 256, 0, stream>>>(hW2, c2, z, pooled + 2 * BATCH * HID);

  cvec_kernel<<<BATCH, HID, 0, stream>>>((const float*)(pooled + 2 * BATCH * HID), W3, b3, c3);
  kernelB<<<gB, 256, 0, stream>>>(hW3, c3, z, pooled + 3 * BATCH * HID);

  head_kernel<<<BATCH, HID, 0, stream>>>((const float*)(pooled + 3 * BATCH * HID),
                                         Wc, bc, Wm0, bm0, Wm1, bm1, Wm2, bm2, Wp, bp,
                                         (float*)d_out);
}
